// BiBlock_7361573945476
// MI455X (gfx1250) — compile-verified, matrix operand order checked by reading, not run on hardware
//
#include <hip/hip_runtime.h>
#include <hip/hip_bf16.h>
#include <math.h>

// ---------------------------------------------------------------- types
typedef __attribute__((ext_vector_type(16))) __bf16 bf16x16;
typedef __attribute__((ext_vector_type(8)))  __bf16 bf16x8;
typedef __attribute__((ext_vector_type(8)))  float  f32x8;

#define DIM   256
#define HH    56
#define WW    56
#define NIMG  16
#define HWSZ  (HH*WW)            // 3136
#define MROWS (NIMG*HWSZ)        // 50176
#define NWIN  49                 // 7x7 windows
#define TOPK  4
#define SCALE 0.0625f            // 256^-0.5

__device__ __forceinline__ bf16x16 mk_frag(const __bf16* p0, const __bf16* p1) {
    bf16x8 lo = *reinterpret_cast<const bf16x8*>(p0);
    bf16x8 hi = *reinterpret_cast<const bf16x8*>(p1);
    bf16x16 r;
#pragma unroll
    for (int i = 0; i < 8; ++i) { r[i] = lo[i]; r[i + 8] = hi[i]; }
    return r;
}
__device__ __forceinline__ f32x8 zero8() {
    f32x8 z = {0.f,0.f,0.f,0.f,0.f,0.f,0.f,0.f};
    return z;
}

// async 16-byte global -> LDS copy (CDNA5, tracked with ASYNCcnt)
__device__ __forceinline__ void async_cp16(unsigned lds_off, const __bf16* g) {
    asm volatile("global_load_async_to_lds_b128 %0, %1, off"
                 :: "v"(lds_off), "v"((unsigned long long)g)
                 : "memory");
}
__device__ __forceinline__ void wait_async_le4() {
    asm volatile("s_wait_asynccnt 0x4" ::: "memory");
}
__device__ __forceinline__ void wait_async_0() {
    asm volatile("s_wait_asynccnt 0x0" ::: "memory");
}

// ---------------------------------------------------------------- fp32 -> bf16 weight convert
__global__ void f2bf_kernel(const float* __restrict__ in, __bf16* __restrict__ out, int n) {
    int i = blockIdx.x * 256 + threadIdx.x;
    if (i < n) out[i] = (__bf16)in[i];
}

// ---------------------------------------------------------------- NCHW -> NHWC + 3x3 dw conv residual
__global__ __launch_bounds__(256) void posconv_kernel(const float* __restrict__ x,
                                                      const float* __restrict__ pw,
                                                      const float* __restrict__ pb,
                                                      float* __restrict__ xh) {
    int b = blockIdx.x;                 // n*HW + h*W + w
    int c = threadIdx.x;
    int w = b % WW, h = (b / WW) % HH, n = b / HWSZ;
    size_t base = ((size_t)n * DIM + c) * HWSZ;
    float ctr = x[base + h * WW + w];
    float acc = ctr + pb[c];
#pragma unroll
    for (int dy = 0; dy < 3; ++dy) {
        int hh = h + dy - 1;
        if (hh < 0 || hh >= HH) continue;
#pragma unroll
        for (int dx = 0; dx < 3; ++dx) {
            int ww2 = w + dx - 1;
            if (ww2 < 0 || ww2 >= WW) continue;
            acc += x[base + hh * WW + ww2] * pw[(dy * 3 + dx) * DIM + c];
        }
    }
    xh[(size_t)b * DIM + c] = acc;
}

// ---------------------------------------------------------------- LayerNorm(256) -> bf16
__global__ __launch_bounds__(256) void ln_kernel(const float* __restrict__ in,
                                                 const float* __restrict__ g,
                                                 const float* __restrict__ bta,
                                                 __bf16* __restrict__ out) {
    int wave = threadIdx.x >> 5, lane = threadIdx.x & 31;
    int row = blockIdx.x * 8 + wave;
    size_t base = (size_t)row * DIM;
    float v[8], s = 0.f, s2 = 0.f;
#pragma unroll
    for (int i = 0; i < 8; ++i) {
        v[i] = in[base + lane + 32 * i];
        s += v[i]; s2 += v[i] * v[i];
    }
#pragma unroll
    for (int m = 1; m < 32; m <<= 1) {
        s  += __shfl_xor(s,  m, 32);
        s2 += __shfl_xor(s2, m, 32);
    }
    float mu = s * (1.f / DIM);
    float var = s2 * (1.f / DIM) - mu * mu;
    float rs = rsqrtf(var + 1e-6f);
#pragma unroll
    for (int i = 0; i < 8; ++i) {
        int c = lane + 32 * i;
        out[base + c] = (__bf16)((v[i] - mu) * rs * g[c] + bta[c]);
    }
}

// ---------------------------------------------------------------- bf16 WMMA GEMM: out = act(A @ W^T + bias) [+res]
// Double-buffered async global->LDS pipeline (ASYNCcnt), 8 WMMAs per K-step.
// OUT_MODE: 0 = bf16 row-major, 1 = f32 row-major (+res), 2 = f32 NCHW-transposed (+res)
template<int OUT_MODE, bool GELU_ACT>
__global__ __launch_bounds__(256) void gemm_kernel(const __bf16* __restrict__ A,
                                                   const __bf16* __restrict__ W,
                                                   const float* __restrict__ bias,
                                                   const float* __restrict__ res,
                                                   void* __restrict__ out,
                                                   int M, int N, int K) {
    __shared__ __bf16 As[2][128 * 40];
    __shared__ __bf16 Ws[2][128 * 40];
    const int tid = threadIdx.x;
    const int n0 = blockIdx.x * 128, m0 = blockIdx.y * 128;
    const int wave = tid >> 5, lane = tid & 31;
    const int wm = wave & 3, wn = wave >> 2;           // 4 x 2 wave grid
    const int r = lane & 15, hi = lane >> 4, ko = hi * 8;
    const int lr = tid >> 1, ls = (tid & 1) * 16;

    const __bf16* agBase = A + (size_t)(m0 + lr) * K + ls;
    const __bf16* wgBase = W + (size_t)(n0 + lr) * K + ls;
    unsigned laA[2], laW[2];
#pragma unroll
    for (int bfi = 0; bfi < 2; ++bfi) {
        laA[bfi] = (unsigned)(size_t)&As[bfi][lr * 40 + ls];
        laW[bfi] = (unsigned)(size_t)&Ws[bfi][lr * 40 + ls];
    }

    f32x8 acc[2][4];
#pragma unroll
    for (int i = 0; i < 2; ++i)
#pragma unroll
        for (int j = 0; j < 4; ++j) acc[i][j] = zero8();

    const int NK = K >> 5;
    // prologue: issue tile 0 into buffer 0 (4 async copies per thread)
    async_cp16(laA[0],      agBase);
    async_cp16(laA[0] + 16, agBase + 8);
    async_cp16(laW[0],      wgBase);
    async_cp16(laW[0] + 16, wgBase + 8);

    for (int kt = 0; kt < NK; ++kt) {
        const int cur = kt & 1;
        if (kt + 1 < NK) {
            const __bf16* ag = agBase + ((kt + 1) << 5);
            const __bf16* wg = wgBase + ((kt + 1) << 5);
            async_cp16(laA[cur ^ 1],      ag);
            async_cp16(laA[cur ^ 1] + 16, ag + 8);
            async_cp16(laW[cur ^ 1],      wg);
            async_cp16(laW[cur ^ 1] + 16, wg + 8);
            wait_async_le4();      // current buffer's 4 copies retired (in-order)
        } else {
            wait_async_0();
        }
        __syncthreads();           // all waves' copies into `cur` visible

        bf16x16 af[2], wf[4];
#pragma unroll
        for (int i = 0; i < 2; ++i) {
            const __bf16* p = &As[cur][(wm * 32 + i * 16 + r) * 40 + ko];
            af[i] = mk_frag(p, p + 16);
        }
#pragma unroll
        for (int j = 0; j < 4; ++j) {
            const __bf16* p = &Ws[cur][(wn * 64 + j * 16 + r) * 40 + ko];
            wf[j] = mk_frag(p, p + 16);
        }
#pragma unroll
        for (int i = 0; i < 2; ++i)
#pragma unroll
            for (int j = 0; j < 4; ++j)
                acc[i][j] = __builtin_amdgcn_wmma_f32_16x16x32_bf16(
                    false, af[i], false, wf[j], (short)0, acc[i][j], false, false);
        __syncthreads();           // reads of `cur` done before it is refilled
    }
    // epilogue
#pragma unroll
    for (int i = 0; i < 2; ++i) {
#pragma unroll
        for (int j = 0; j < 4; ++j) {
            int mbase = m0 + wm * 32 + i * 16 + hi * 8;
            int col = n0 + wn * 64 + j * 16 + r;
            float bv = bias[col];
#pragma unroll
            for (int v = 0; v < 8; ++v) {
                int row = mbase + v;
                float val = acc[i][j][v] + bv;
                if (GELU_ACT) val = 0.5f * val * (1.f + erff(val * 0.70710678118f));
                if (OUT_MODE == 1 || OUT_MODE == 2) val += res[(size_t)row * N + col];
                if (OUT_MODE == 0) {
                    ((__bf16*)out)[(size_t)row * N + col] = (__bf16)val;
                } else if (OUT_MODE == 1) {
                    ((float*)out)[(size_t)row * N + col] = val;
                } else {
                    int nImg = row / HWSZ, hw = row % HWSZ;
                    ((float*)out)[((size_t)nImg * DIM + col) * HWSZ + hw] = val;
                }
            }
        }
    }
}

// ---------------------------------------------------------------- window pooling: q_win, k_win (f32), kv_pix (bf16)
__global__ __launch_bounds__(256) void pool_kernel(const __bf16* __restrict__ qkv,
                                                   float* __restrict__ qwin,
                                                   float* __restrict__ kwin,
                                                   __bf16* __restrict__ kvpix) {
    int b = blockIdx.x;                  // n*49 + p
    int n = b / NWIN, p = b % NWIN;
    int wy = p / 7, wx = p % 7;
    int c = threadIdx.x;
    float qs = 0.f, ks = 0.f;
#pragma unroll
    for (int ry = 0; ry < 4; ++ry) {
#pragma unroll
        for (int rx = 0; rx < 4; ++rx) {
            float kr = 0.f, vr = 0.f;
#pragma unroll
            for (int iy = 0; iy < 2; ++iy) {
#pragma unroll
                for (int ix = 0; ix < 2; ++ix) {
                    int h = wy * 8 + ry * 2 + iy;
                    int w = wx * 8 + rx * 2 + ix;
                    size_t row = ((size_t)(n * HH + h) * WW + w) * 768;
                    float q = (float)qkv[row + c];
                    float k = (float)qkv[row + 256 + c];
                    float v = (float)qkv[row + 512 + c];
                    qs += q; ks += k; kr += k; vr += v;
                }
            }
            size_t po = ((size_t)b * 16 + ry * 4 + rx) * 512;
            kvpix[po + c]       = (__bf16)(kr * 0.25f);
            kvpix[po + 256 + c] = (__bf16)(vr * 0.25f);
        }
    }
    qwin[(size_t)b * DIM + c] = qs * (1.f / 64.f);
    kwin[(size_t)b * DIM + c] = ks * (1.f / 64.f);
}

// ---------------------------------------------------------------- routing logits (49x49 per image)
__global__ __launch_bounds__(256) void logits_kernel(const float* __restrict__ qwin,
                                                     const float* __restrict__ kwin,
                                                     float* __restrict__ logits) {
    int n = blockIdx.x;
    for (int pq = threadIdx.x; pq < NWIN * NWIN; pq += 256) {
        int p = pq / NWIN, q = pq % NWIN;
        const float* qp = qwin + ((size_t)n * NWIN + p) * DIM;
        const float* kp = kwin + ((size_t)n * NWIN + q) * DIM;
        float d = 0.f;
        for (int c = 0; c < DIM; ++c) d += qp[c] * kp[c];
        logits[((size_t)n * NWIN + p) * NWIN + q] = d * SCALE;
    }
}

// ---------------------------------------------------------------- top-4 (stable, descending; lower index on ties)
__global__ void topk_kernel(const float* __restrict__ logits, int* __restrict__ idx) {
    int t = blockIdx.x * 256 + threadIdx.x;
    if (t >= NIMG * NWIN) return;
    const float* l = logits + (size_t)t * NWIN;
    int chosen[TOPK];
#pragma unroll
    for (int j = 0; j < TOPK; ++j) {
        float best = -3.4e38f; int bi = 0;
        for (int q = 0; q < NWIN; ++q) {
            bool skip = false;
            for (int jj = 0; jj < j; ++jj) if (chosen[jj] == q) skip = true;
            if (skip) continue;
            float v = l[q];
            if (v > best) { best = v; bi = q; }
        }
        chosen[j] = bi;
        idx[(size_t)t * TOPK + j] = bi;
    }
}

// ---------------------------------------------------------------- routed attention (one block per (n,window), one wave per head)
__global__ __launch_bounds__(256) void attn_kernel(const __bf16* __restrict__ qkv,
                                                   const __bf16* __restrict__ kvpix,
                                                   const int* __restrict__ idx,
                                                   __bf16* __restrict__ wo_in) {
    __shared__ __bf16 vT[256 * 64];          // [channel][key], 32 KB
    __shared__ __bf16 attb[8 * 64 * 64];     // per-wave att, 64 KB
    int b = blockIdx.x;
    int n = b / NWIN, p = b % NWIN;
    int wy = p / 7, wx = p % 7;
    int tid = threadIdx.x;
    int ix4[TOPK];
#pragma unroll
    for (int j = 0; j < TOPK; ++j) ix4[j] = idx[(size_t)b * TOPK + j];

    // build vT (channel-major V of the 64 selected keys)
    {
        int c = tid;
#pragma unroll
        for (int key = 0; key < 64; ++key) {
            int j = key >> 4, rr = key & 15;
            size_t off = ((size_t)(n * NWIN + ix4[j]) * 16 + rr) * 512 + 256 + c;
            vT[c * 64 + key] = kvpix[off];
        }
    }
    __syncthreads();

    int wave = tid >> 5, lane = tid & 31;
    int head = wave;
    int r = lane & 15, hi = lane >> 4, ko = hi * 8;
    __bf16* attw = &attb[wave * 64 * 64];

    // B fragments for QK^T: column = key kt*16+r, k = channel chunks
    bf16x16 bq[4];
#pragma unroll
    for (int kt = 0; kt < 4; ++kt) {
        const __bf16* kp = kvpix + ((size_t)(n * NWIN + ix4[kt]) * 16 + r) * 512 + head * 32 + ko;
        bq[kt] = mk_frag(kp, kp + 16);
    }

#pragma unroll
    for (int mt = 0; mt < 4; ++mt) {
        int m = mt * 16 + r;                       // pixel row
        int h = wy * 8 + (m >> 3), w = wx * 8 + (m & 7);
        const __bf16* qp = qkv + ((size_t)(n * HH + h) * WW + w) * 768 + head * 32 + ko;
        bf16x16 af = mk_frag(qp, qp + 16);
        f32x8 acc[4];
#pragma unroll
        for (int kt = 0; kt < 4; ++kt) acc[kt] = zero8();
#pragma unroll
        for (int kt = 0; kt < 4; ++kt)
            acc[kt] = __builtin_amdgcn_wmma_f32_16x16x32_bf16(
                false, af, false, bq[kt], (short)0, acc[kt], false, false);
        // softmax per row (rows live at fixed vgpr slot across the 16-lane half)
#pragma unroll
        for (int v = 0; v < 8; ++v) {
            float s[4];
#pragma unroll
            for (int kt = 0; kt < 4; ++kt) s[kt] = acc[kt][v] * SCALE;
            float mx = fmaxf(fmaxf(s[0], s[1]), fmaxf(s[2], s[3]));
#pragma unroll
            for (int msk = 1; msk < 16; msk <<= 1) mx = fmaxf(mx, __shfl_xor(mx, msk, 32));
            float e[4], sum = 0.f;
#pragma unroll
            for (int kt = 0; kt < 4; ++kt) { e[kt] = __expf(s[kt] - mx); sum += e[kt]; }
#pragma unroll
            for (int msk = 1; msk < 16; msk <<= 1) sum += __shfl_xor(sum, msk, 32);
            float rinv = 1.f / sum;
            int mrow = mt * 16 + v + hi * 8;
#pragma unroll
            for (int kt = 0; kt < 4; ++kt)
                attw[mrow * 64 + kt * 16 + r] = (__bf16)(e[kt] * rinv);
        }
    }
    __syncthreads();

    // out = att @ V  (M=64 pixels, K=64 keys, N=32 head channels)
#pragma unroll
    for (int mt = 0; mt < 4; ++mt) {
#pragma unroll
        for (int nt = 0; nt < 2; ++nt) {
            f32x8 o = zero8();
#pragma unroll
            for (int kk = 0; kk < 2; ++kk) {
                const __bf16* ap = &attw[(mt * 16 + r) * 64 + kk * 32 + ko];
                bf16x16 aa = mk_frag(ap, ap + 16);
                const __bf16* bp = &vT[(head * 32 + nt * 16 + r) * 64 + kk * 32 + ko];
                bf16x16 bb = mk_frag(bp, bp + 16);
                o = __builtin_amdgcn_wmma_f32_16x16x32_bf16(
                    false, aa, false, bb, (short)0, o, false, false);
            }
#pragma unroll
            for (int v = 0; v < 8; ++v) {
                int m = mt * 16 + v + hi * 8;
                int h = wy * 8 + (m >> 3), w = wx * 8 + (m & 7);
                wo_in[((size_t)(n * HH + h) * WW + w) * DIM + head * 32 + nt * 16 + r] = (__bf16)o[v];
            }
        }
    }
}

// ---------------------------------------------------------------- LePE: wo_in += dwconv5x5(v_img)
__global__ __launch_bounds__(256) void lepe_kernel(const __bf16* __restrict__ qkv,
                                                   const float* __restrict__ lw,
                                                   const float* __restrict__ lb,
                                                   __bf16* __restrict__ wo_in) {
    int b = blockIdx.x;
    int c = threadIdx.x;
    int w = b % WW, h = (b / WW) % HH, n = b / HWSZ;
    float acc = lb[c];
#pragma unroll
    for (int dy = 0; dy < 5; ++dy) {
        int hh = h + dy - 2;
        if (hh < 0 || hh >= HH) continue;
#pragma unroll
        for (int dx = 0; dx < 5; ++dx) {
            int ww2 = w + dx - 2;
            if (ww2 < 0 || ww2 >= WW) continue;
            float v = (float)qkv[((size_t)(n * HH + hh) * WW + ww2) * 768 + 512 + c];
            acc += v * lw[(dy * 5 + dx) * DIM + c];
        }
    }
    size_t o = (size_t)b * DIM + c;
    wo_in[o] = (__bf16)((float)wo_in[o] + acc);
}

// ---------------------------------------------------------------- host
extern "C" void kernel_launch(void* const* d_in, const int* in_sizes, int n_in,
                              void* d_out, int out_size, void* d_ws, size_t ws_size,
                              hipStream_t stream) {
    const float* x      = (const float*)d_in[0];
    const float* pos_w  = (const float*)d_in[1];
    const float* pos_b  = (const float*)d_in[2];
    const float* ln1_g  = (const float*)d_in[3];
    const float* ln1_b  = (const float*)d_in[4];
    const float* ln2_g  = (const float*)d_in[5];
    const float* ln2_b  = (const float*)d_in[6];
    const float* qkv_w  = (const float*)d_in[7];
    const float* qkv_b  = (const float*)d_in[8];
    const float* wo_w   = (const float*)d_in[9];
    const float* wo_b   = (const float*)d_in[10];
    const float* lepe_w = (const float*)d_in[11];
    const float* lepe_b = (const float*)d_in[12];
    const float* fc1_w  = (const float*)d_in[13];
    const float* fc1_b  = (const float*)d_in[14];
    const float* fc2_w  = (const float*)d_in[15];
    const float* fc2_b  = (const float*)d_in[16];

    char* ws = (char*)d_ws;
    // workspace layout (bytes)
    float*  xh     = (float*) (ws + 0);                       //  51,380,224
    __bf16* lnbuf  = (__bf16*)(ws + 51380224);                //  25,690,112
    __bf16* wo_in  = (__bf16*)(ws + 77070336);                //  25,690,112
    __bf16* wbf    = (__bf16*)(ws + 102760448);               //   1,572,864
    float*  qwin   = (float*) (ws + 104333312);               //     802,816
    float*  kwin   = (float*) (ws + 105136128);               //     802,816
    float*  logits = (float*) (ws + 105938944);               //     153,856
    int*    idx    = (int*)   (ws + 106092800);               //      12,544
    char*   big    = ws + 106105344;                          // 102,760,448 (qkv+kvpix, later h1)
    __bf16* qkvbf  = (__bf16*)(big);                          //  77,070,336
    __bf16* kvpix  = (__bf16*)(big + 77070336);               //  12,845,056
    __bf16* h1     = (__bf16*)(big);                          // 102,760,448 (reuses qkv region)

    __bf16* qkvw_bf = wbf;                       // 768*256
    __bf16* wow_bf  = wbf + 196608;              // 256*256
    __bf16* fc1w_bf = wbf + 262144;              // 1024*256
    __bf16* fc2w_bf = wbf + 524288;              // 256*1024

    // 1. weights -> bf16
    f2bf_kernel<<<(196608 + 255) / 256, 256, 0, stream>>>(qkv_w, qkvw_bf, 196608);
    f2bf_kernel<<<(65536  + 255) / 256, 256, 0, stream>>>(wo_w,  wow_bf,  65536);
    f2bf_kernel<<<(262144 + 255) / 256, 256, 0, stream>>>(fc1_w, fc1w_bf, 262144);
    f2bf_kernel<<<(262144 + 255) / 256, 256, 0, stream>>>(fc2_w, fc2w_bf, 262144);

    // 2. xh = NHWC(x) + dwconv3x3
    posconv_kernel<<<MROWS, 256, 0, stream>>>(x, pos_w, pos_b, xh);
    // 3. LN1 -> bf16
    ln_kernel<<<MROWS / 8, 256, 0, stream>>>(xh, ln1_g, ln1_b, lnbuf);
    // 4. qkv GEMM (bf16 out)
    gemm_kernel<0, false><<<dim3(768 / 128, MROWS / 128), 256, 0, stream>>>(
        lnbuf, qkvw_bf, qkv_b, nullptr, qkvbf, MROWS, 768, 256);
    // 5. window pooling
    pool_kernel<<<NIMG * NWIN, 256, 0, stream>>>(qkvbf, qwin, kwin, kvpix);
    // 6. routing logits + 7. top-4
    logits_kernel<<<NIMG, 256, 0, stream>>>(qwin, kwin, logits);
    topk_kernel<<<(NIMG * NWIN + 255) / 256, 256, 0, stream>>>(logits, idx);
    // 8. routed attention -> wo_in
    attn_kernel<<<NIMG * NWIN, 256, 0, stream>>>(qkvbf, kvpix, idx, wo_in);
    // 9. LePE add
    lepe_kernel<<<MROWS, 256, 0, stream>>>(qkvbf, lepe_w, lepe_b, wo_in);
    // 10. wo GEMM (+xh residual, fp32, in-place into xh)
    gemm_kernel<1, false><<<dim3(256 / 128, MROWS / 128), 256, 0, stream>>>(
        wo_in, wow_bf, wo_b, xh, xh, MROWS, 256, 256);
    // 11. LN2 -> bf16
    ln_kernel<<<MROWS / 8, 256, 0, stream>>>(xh, ln2_g, ln2_b, lnbuf);
    // 12. fc1 GEMM + GELU (bf16 out, reuses qkv region)
    gemm_kernel<0, true><<<dim3(1024 / 128, MROWS / 128), 256, 0, stream>>>(
        lnbuf, fc1w_bf, fc1_b, nullptr, h1, MROWS, 1024, 256);
    // 13. fc2 GEMM + residual + NHWC->NCHW transpose into d_out
    gemm_kernel<2, false><<<dim3(256 / 128, MROWS / 128), 256, 0, stream>>>(
        h1, fc2w_bf, fc2_b, xh, d_out, MROWS, 256, 1024);
}
